// NanoChatLM_26328149524656
// MI455X (gfx1250) — matrix operand order repaired and checked by reading, not compiled, not run on hardware
//
#include <hip/hip_runtime.h>
#include <math.h>

typedef __attribute__((ext_vector_type(16))) _Float16 v16h;
typedef __attribute__((ext_vector_type(8)))  _Float16 v8h;
typedef __attribute__((ext_vector_type(8)))  float    v8f;
typedef int v4i __attribute__((vector_size(16)));

#define LDS_ALIGN __attribute__((aligned(32)))

// ---- model dims (compile-time) ----
static constexpr int Lc  = 6;
static constexpr int Dc  = 768;
static constexpr int Hc  = 12;
static constexpr int Vc  = 32000;
static constexpr int Tc  = 1024;
static constexpr int Bc  = 4;
static constexpr int HDc = Dc / Hc;   // 64
static constexpr int FFc = 4 * Dc;    // 3072
static constexpr int Mc  = Bc * Tc;   // 4096 activation rows

// ---------------------------------------------------------------------------
// Async global->LDS copy (gfx1250 GLOBAL_LOAD_ASYNC_TO_LDS_B128, ASYNCcnt),
// with plain-copy fallback so the file compiles on any toolchain.
// ---------------------------------------------------------------------------
#if __has_builtin(__builtin_amdgcn_global_load_async_to_lds_b128)
#define HAVE_ASYNC_LDS 1
#endif

__device__ __forceinline__ void copy16B(const _Float16* g, _Float16* l) {
#ifdef HAVE_ASYNC_LDS
  __builtin_amdgcn_global_load_async_to_lds_b128((v4i*)g, (v4i*)l, 0, 0);
#else
  *(v8h*)l = *(const v8h*)g;
#endif
}

__device__ __forceinline__ void wait_async_lds() {
#ifdef HAVE_ASYNC_LDS
#if __has_builtin(__builtin_amdgcn_s_wait_asynccnt)
  __builtin_amdgcn_s_wait_asynccnt(0);
#else
  asm volatile("s_wait_asynccnt 0x0" ::: "memory");
#endif
#endif
}

// ---------------------------------------------------------------------------
// WMMA helpers (gfx1250 wave32 fragment layouts, CDNA5 ISA 7.12.2)
// ---------------------------------------------------------------------------
__device__ __forceinline__ v8f wmma32(v16h a, v16h b, v8f c) {
  return __builtin_amdgcn_wmma_f32_16x16x32_f16(false, a, false, b, (short)0, c,
                                                false, false);
}

// A matrix 16x32 f16 from LDS [m][k] (row stride rs halfs)
__device__ __forceinline__ v16h load_a_frag(const _Float16* base, int rs) {
  const int lane = threadIdx.x & 31;
  const int m = lane & 15, kg = lane >> 4;
  const _Float16* p = base + m * rs + kg * 16;
  v8h lo = *(const v8h*)(p);
  v8h hi = *(const v8h*)(p + 8);
  v16h a;
#pragma unroll
  for (int i = 0; i < 8; ++i) { a[i] = lo[i]; a[i + 8] = hi[i]; }
  return a;
}

// B matrix 32x16 f16 from LDS [n][k] (row stride rs halfs) - contiguous 32B
__device__ __forceinline__ v16h load_b_frag(const _Float16* base, int rs) {
  const int lane = threadIdx.x & 31;
  const int n = lane & 15, kg = lane >> 4;
  return *(const v16h*)(base + n * rs + kg * 16);
}

// ---------------------------------------------------------------------------
// One-time per-launch conversions
// ---------------------------------------------------------------------------
// [K,N] f32 -> [N,K] f16 (weights into the B-fragment-native layout)
__global__ __launch_bounds__(256) void transpose_cvt_k(const float* __restrict__ in,
                                                       _Float16* __restrict__ out,
                                                       int K, int N) {
  __shared__ float tile[32][33];
  const int kb = blockIdx.y * 32, nb = blockIdx.x * 32;
  const int tx = threadIdx.x & 31, ty = threadIdx.x >> 5;  // 32 x 8
#pragma unroll
  for (int i = 0; i < 4; ++i)
    tile[ty + 8 * i][tx] = in[(size_t)(kb + ty + 8 * i) * N + nb + tx];
  __syncthreads();
#pragma unroll
  for (int i = 0; i < 4; ++i)
    out[(size_t)(nb + ty + 8 * i) * K + kb + tx] = (_Float16)tile[tx][ty + 8 * i];
}

// plain f32 -> f16 convert (Wtok is already [V][D] == [N][K])
__global__ __launch_bounds__(256) void cvt_k(const float* __restrict__ in,
                                             _Float16* __restrict__ out) {
  const int i = (blockIdx.x * 256 + threadIdx.x) * 4;
  float4 f = *(const float4*)(in + i);
  out[i + 0] = (_Float16)f.x;
  out[i + 1] = (_Float16)f.y;
  out[i + 2] = (_Float16)f.z;
  out[i + 3] = (_Float16)f.w;
}

// ---------------------------------------------------------------------------
// Embedding: x = Wtok[idx] + Wpos[t]; v = 0
// ---------------------------------------------------------------------------
__global__ __launch_bounds__(256) void embed_k(const int* __restrict__ idx,
                                               const float* __restrict__ Wtok,
                                               const float* __restrict__ Wpos,
                                               float* __restrict__ x,
                                               float* __restrict__ v) {
  const int row = blockIdx.x;  // b*Tc + t
  const int t = row % Tc;
  const int tok = idx[row];
  for (int j = threadIdx.x; j < Dc; j += 256) {
    x[(size_t)row * Dc + j] = Wtok[(size_t)tok * Dc + j] + Wpos[(size_t)t * Dc + j];
    v[(size_t)row * Dc + j] = 0.0f;
  }
}

// ---------------------------------------------------------------------------
// LayerNorm (f32 in, f16 out): one block per D=768 row
// ---------------------------------------------------------------------------
__global__ __launch_bounds__(256) void layernorm_k(const float* __restrict__ x,
                                                   const float* __restrict__ g,
                                                   const float* __restrict__ bt,
                                                   _Float16* __restrict__ out) {
  __shared__ float r1[8], r2[8];
  const int row = blockIdx.x, tid = threadIdx.x;
  const float* xr = x + (size_t)row * Dc;
  float v0 = xr[tid], v1 = xr[tid + 256], v2 = xr[tid + 512];
  float s1 = v0 + v1 + v2;
  float s2 = v0 * v0 + v1 * v1 + v2 * v2;
#pragma unroll
  for (int off = 16; off; off >>= 1) {
    s1 += __shfl_xor(s1, off, 32);
    s2 += __shfl_xor(s2, off, 32);
  }
  if ((tid & 31) == 0) { r1[tid >> 5] = s1; r2[tid >> 5] = s2; }
  __syncthreads();
  s1 = 0.0f; s2 = 0.0f;
#pragma unroll
  for (int i = 0; i < 8; ++i) { s1 += r1[i]; s2 += r2[i]; }
  const float mu = s1 / (float)Dc;
  const float var = s2 / (float)Dc - mu * mu;
  const float rstd = rsqrtf(var + 1e-5f);
  _Float16* orow = out + (size_t)row * Dc;
  orow[tid]       = (_Float16)((v0 - mu) * rstd * g[tid]       + bt[tid]);
  orow[tid + 256] = (_Float16)((v1 - mu) * rstd * g[tid + 256] + bt[tid + 256]);
  orow[tid + 512] = (_Float16)((v2 - mu) * rstd * g[tid + 512] + bt[tid + 512]);
}

// ---------------------------------------------------------------------------
// GEMM: out[M,N] = act(A[M,K] @ Bt^T + bias); A f16 [M][K], Bt f16 [N][K].
// 128x128x32 tile, double-buffered LDS, async global->LDS staging.
// 8 waves (4m x 2n), each 2x4 WMMA 16x16 tiles, f32 accumulate.
// ---------------------------------------------------------------------------
enum { ACT_NONE = 0, ACT_GELU = 1 };

__device__ __forceinline__ void stage_tile(const _Float16* __restrict__ g, int ldg,
                                           int row0, int k0, _Float16* l) {
  const int tid = threadIdx.x;
#pragma unroll
  for (int i = 0; i < 2; ++i) {
    const int e = i * 256 + tid;       // 0..511 : 128 rows x 4 x 16B
    const int r = e >> 2, q = e & 3;
    copy16B(g + (size_t)(row0 + r) * ldg + k0 + q * 8, l + r * 32 + q * 8);
  }
}

template <int ACT, bool ACCUM, bool OUTH>
__global__ __launch_bounds__(256) void gemm_wmma(const _Float16* __restrict__ A,
                                                 const _Float16* __restrict__ Bt,
                                                 const float* __restrict__ bias,
                                                 float* __restrict__ C,
                                                 _Float16* __restrict__ Ch,
                                                 int M, int N, int K) {
  __shared__ LDS_ALIGN _Float16 As[2][128 * 32];
  __shared__ LDS_ALIGN _Float16 Bs[2][128 * 32];

  const int tid = threadIdx.x;
  const int lane = tid & 31;
  const int wid = tid >> 5;
  const int wm = wid & 3;    // 32-row strip
  const int wn = wid >> 2;   // 64-col strip
  const int bn0 = blockIdx.x * 128;
  const int bm0 = blockIdx.y * 128;
  const int hi = lane >> 4, lo = lane & 15;

  v8f acc[2][4];
#pragma unroll
  for (int i = 0; i < 2; ++i)
#pragma unroll
    for (int j = 0; j < 4; ++j)
#pragma unroll
      for (int e = 0; e < 8; ++e) acc[i][j][e] = 0.0f;

  const int nk = K >> 5;
  stage_tile(A, K, bm0, 0, As[0]);
  stage_tile(Bt, K, bn0, 0, Bs[0]);
  wait_async_lds();
  __syncthreads();

  for (int i = 0; i < nk; ++i) {
    const int cur = i & 1, nxt = cur ^ 1;
    if (i + 1 < nk) {
      stage_tile(A, K, bm0, (i + 1) * 32, As[nxt]);
      stage_tile(Bt, K, bn0, (i + 1) * 32, Bs[nxt]);
    }
    v16h af[2], bf[4];
#pragma unroll
    for (int mt = 0; mt < 2; ++mt)
      af[mt] = load_a_frag(As[cur] + (wm * 32 + mt * 16) * 32, 32);
#pragma unroll
    for (int nt = 0; nt < 4; ++nt)
      bf[nt] = load_b_frag(Bs[cur] + (wn * 64 + nt * 16) * 32, 32);
#pragma unroll
    for (int mt = 0; mt < 2; ++mt)
#pragma unroll
      for (int nt = 0; nt < 4; ++nt)
        acc[mt][nt] = wmma32(af[mt], bf[nt], acc[mt][nt]);
    wait_async_lds();
    __syncthreads();
  }

  // epilogue: lane lo = n, VGPR r -> m = r + 8*hi
#pragma unroll
  for (int mt = 0; mt < 2; ++mt)
#pragma unroll
    for (int nt = 0; nt < 4; ++nt)
#pragma unroll
      for (int r = 0; r < 8; ++r) {
        const int gm = bm0 + wm * 32 + mt * 16 + r + 8 * hi;
        const int gn = bn0 + wn * 64 + nt * 16 + lo;
        float val = acc[mt][nt][r];
        if (bias) val += bias[gn];
        if (ACT == ACT_GELU)
          val = 0.5f * val * (1.0f + erff(val * 0.70710678118654752f));
        const size_t off = (size_t)gm * N + gn;
        if (OUTH) Ch[off] = (_Float16)val;
        else if (ACCUM) C[off] += val;
        else C[off] = val;
      }
}

// ---------------------------------------------------------------------------
// Flash attention on f16 qkv: one block per (b, h, 128-row q tile).
// ---------------------------------------------------------------------------
__global__ __launch_bounds__(256) void attn_flash(const _Float16* __restrict__ qkvh,
                                                  _Float16* __restrict__ y) {
  __shared__ LDS_ALIGN _Float16 Qs[128 * 64];     // [m][hd]
  __shared__ LDS_ALIGN _Float16 Ks[64 * 64];      // [kv][hd]  (B [n][k] for QK^T)
  __shared__ LDS_ALIGN _Float16 Vs[64 * 64];      // [hd][kv]  (B [n][k] for PV)
  __shared__ LDS_ALIGN _Float16 Ps[8 * 16 * 64];  // per-wave P tile [m][kv]

  const int tid = threadIdx.x, lane = tid & 31, wid = tid >> 5;
  const int hi = lane >> 4, lo = lane & 15;
  const int nqt = Tc / 128;  // 8
  const int qt = blockIdx.x % nqt;
  const int h  = (blockIdx.x / nqt) % Hc;
  const int b  = blockIdx.x / (nqt * Hc);
  const int q0 = qt * 128;
  const size_t rstride = (size_t)(3 * Dc);
  const float scale = 0.125f;  // 1/sqrt(64)

  // load Q tile (async: 1024 x 16B)
#pragma unroll
  for (int i = 0; i < 4; ++i) {
    const int e = i * 256 + tid;
    const int m = e >> 3, q = e & 7;
    copy16B(qkvh + (size_t)(b * Tc + q0 + m) * rstride + h * HDc + q * 8,
            Qs + m * 64 + q * 8);
  }
  wait_async_lds();
  __syncthreads();

  const int m0 = wid * 16;
  v8f o[4];
  float mx[8], ls[8];
#pragma unroll
  for (int nt = 0; nt < 4; ++nt)
#pragma unroll
    for (int r = 0; r < 8; ++r) o[nt][r] = 0.0f;
#pragma unroll
  for (int r = 0; r < 8; ++r) { mx[r] = -1e30f; ls[r] = 0.0f; }

  for (int c0 = 0; c0 < q0 + 128; c0 += 64) {
    // K chunk: straight async copy; V chunk: transposed (manual halves)
#pragma unroll
    for (int i = 0; i < 2; ++i) {
      const int e = i * 256 + tid;
      const int kv = e >> 3, q = e & 7;
      copy16B(qkvh + (size_t)(b * Tc + c0 + kv) * rstride + Dc + h * HDc + q * 8,
              Ks + kv * 64 + q * 8);
    }
#pragma unroll
    for (int i = 0; i < 16; ++i) {
      const int e = i * 256 + tid;
      const int kv = e >> 6, d = e & 63;
      Vs[d * 64 + kv] =
          qkvh[(size_t)(b * Tc + c0 + kv) * rstride + 2 * Dc + h * HDc + d];
    }
    wait_async_lds();
    __syncthreads();

    // S = Q @ K^T (16 x 64 per wave)
    const v16h aS0 = load_a_frag(Qs + m0 * 64 + 0, 64);
    const v16h aS1 = load_a_frag(Qs + m0 * 64 + 32, 64);
    v8f s[4];
#pragma unroll
    for (int nt = 0; nt < 4; ++nt) {
      v8f z;
#pragma unroll
      for (int e = 0; e < 8; ++e) z[e] = 0.0f;
      z = wmma32(aS0, load_b_frag(Ks + nt * 16 * 64 + 0, 64), z);
      z = wmma32(aS1, load_b_frag(Ks + nt * 16 * 64 + 32, 64), z);
      s[nt] = z;
    }

    // scale, causal mask, online softmax update, stash P as f16
#pragma unroll
    for (int r = 0; r < 8; ++r) {
      const int grow = q0 + m0 + r + 8 * hi;
      float rm = -1e30f;
#pragma unroll
      for (int nt = 0; nt < 4; ++nt) {
        const int gcol = c0 + nt * 16 + lo;
        float sv = s[nt][r] * scale;
        sv = (gcol > grow) ? -1e30f : sv;
        s[nt][r] = sv;
        rm = fmaxf(rm, sv);
      }
#pragma unroll
      for (int off = 1; off < 16; off <<= 1)
        rm = fmaxf(rm, __shfl_xor(rm, off, 32));
      const float mnew = fmaxf(mx[r], rm);
      const float alpha = __expf(mx[r] - mnew);
      float rs = 0.0f;
#pragma unroll
      for (int nt = 0; nt < 4; ++nt) {
        const float p = __expf(s[nt][r] - mnew);
        s[nt][r] = p;
        rs += p;
      }
#pragma unroll
      for (int off = 1; off < 16; off <<= 1) rs += __shfl_xor(rs, off, 32);
      ls[r] = ls[r] * alpha + rs;
      mx[r] = mnew;
#pragma unroll
      for (int nt = 0; nt < 4; ++nt) o[nt][r] *= alpha;
      const int ml = r + 8 * hi;
#pragma unroll
      for (int nt = 0; nt < 4; ++nt)
        Ps[wid * 1024 + ml * 64 + nt * 16 + lo] = (_Float16)s[nt][r];
    }

    // O += P @ V (per-wave private P region; same-wave LDS ops are ordered)
    const v16h aP0 = load_a_frag(Ps + wid * 1024 + 0, 64);
    const v16h aP1 = load_a_frag(Ps + wid * 1024 + 32, 64);
#pragma unroll
    for (int nt = 0; nt < 4; ++nt) {
      o[nt] = wmma32(aP0, load_b_frag(Vs + nt * 16 * 64 + 0, 64), o[nt]);
      o[nt] = wmma32(aP1, load_b_frag(Vs + nt * 16 * 64 + 32, 64), o[nt]);
    }
    __syncthreads();
  }

  // normalize, write y (f16) for the projection GEMM
#pragma unroll
  for (int nt = 0; nt < 4; ++nt)
#pragma unroll
    for (int r = 0; r < 8; ++r) {
      const int gm = q0 + m0 + r + 8 * hi;
      const int d = nt * 16 + lo;
      y[(size_t)(b * Tc + gm) * Dc + h * HDc + d] = (_Float16)(o[nt][r] / ls[r]);
    }
}

// ---------------------------------------------------------------------------
// HORN momentum residual: v = sigmoid(m)*v + softplus(eta)*u; x += v
// ---------------------------------------------------------------------------
__global__ __launch_bounds__(256) void horn_k(const float* __restrict__ u,
                                              const float* __restrict__ mlog,
                                              const float* __restrict__ etap,
                                              float* __restrict__ v,
                                              float* __restrict__ x) {
  const int e = blockIdx.x * 256 + threadIdx.x;
  const int d = e % Dc;
  const float eta = log1pf(__expf(etap[d]));
  const float mm = 1.0f / (1.0f + __expf(-mlog[d]));
  const float vn = mm * v[e] + eta * u[e];
  v[e] = vn;
  x[e] += vn;
}

// ---------------------------------------------------------------------------
// Host orchestration (graph-capture safe: stream launches only)
// ---------------------------------------------------------------------------
extern "C" void kernel_launch(void* const* d_in, const int* in_sizes, int n_in,
                              void* d_out, int out_size, void* d_ws,
                              size_t ws_size, hipStream_t stream) {
  (void)in_sizes; (void)n_in; (void)out_size; (void)ws_size;

  const int*   idx   = (const int*)d_in[0];
  const float* Wtok  = (const float*)d_in[1];
  const float* Wpos  = (const float*)d_in[2];
  const float* ln1g  = (const float*)d_in[3];
  const float* ln1b  = (const float*)d_in[4];
  const float* Wqkv  = (const float*)d_in[5];
  const float* bqkv  = (const float*)d_in[6];
  const float* Wproj = (const float*)d_in[7];
  const float* bproj = (const float*)d_in[8];
  const float* ln2g  = (const float*)d_in[9];
  const float* ln2b  = (const float*)d_in[10];
  const float* Wfc   = (const float*)d_in[11];
  const float* bfc   = (const float*)d_in[12];
  const float* Wfc2  = (const float*)d_in[13];
  const float* bfc2  = (const float*)d_in[14];
  const float* mlog  = (const float*)d_in[15];
  const float* etap  = (const float*)d_in[16];
  const float* lnfg  = (const float*)d_in[17];
  const float* lnfb  = (const float*)d_in[18];
  float* out = (float*)d_out;

  const size_t MD = (size_t)Mc * Dc;
  char* p = (char*)d_ws;
  float* x = (float*)p;        p += MD * 4;
  float* v = (float*)p;        p += MD * 4;
  float* u = (float*)p;        p += MD * 4;
  _Float16* h16   = (_Float16*)p;  p += MD * 2;
  _Float16* qkv16 = (_Float16*)p;  p += 3 * MD * 2;
  _Float16* y16   = (_Float16*)p;  p += MD * 2;
  _Float16* hff16 = (_Float16*)p;  p += 4 * MD * 2;
  _Float16* Wt    = (_Float16*)p;  p += (size_t)Lc * Dc * Dc * 12 * 2;
  _Float16* Wtok16= (_Float16*)p;  p += (size_t)Vc * Dc * 2;

  const dim3 blk(256);
  const int MB = Mc / 128;  // 32 row-blocks

  // one-time (per launch) weight conversion into f16 [N][K] layouts
  cvt_k<<<(Vc * Dc) / 1024, blk, 0, stream>>>(Wtok, Wtok16);
  for (int l = 0; l < Lc; ++l) {
    _Float16* WtL    = Wt + (size_t)l * Dc * Dc * 12;
    _Float16* wqkvT  = WtL;                              // [3D][D]
    _Float16* wprojT = wqkvT + (size_t)3 * Dc * Dc;      // [D][D]
    _Float16* wfcT   = wprojT + (size_t)Dc * Dc;         // [FF][D]
    _Float16* wfc2T  = wfcT + (size_t)FFc * Dc;          // [D][FF]
    transpose_cvt_k<<<dim3(3 * Dc / 32, Dc / 32), blk, 0, stream>>>(
        Wqkv + (size_t)l * Dc * 3 * Dc, wqkvT, Dc, 3 * Dc);
    transpose_cvt_k<<<dim3(Dc / 32, Dc / 32), blk, 0, stream>>>(
        Wproj + (size_t)l * Dc * Dc, wprojT, Dc, Dc);
    transpose_cvt_k<<<dim3(FFc / 32, Dc / 32), blk, 0, stream>>>(
        Wfc + (size_t)l * Dc * FFc, wfcT, Dc, FFc);
    transpose_cvt_k<<<dim3(Dc / 32, FFc / 32), blk, 0, stream>>>(
        Wfc2 + (size_t)l * FFc * Dc, wfc2T, FFc, Dc);
  }

  embed_k<<<Mc, blk, 0, stream>>>(idx, Wtok, Wpos, x, v);

  for (int l = 0; l < Lc; ++l) {
    _Float16* WtL    = Wt + (size_t)l * Dc * Dc * 12;
    _Float16* wqkvT  = WtL;
    _Float16* wprojT = wqkvT + (size_t)3 * Dc * Dc;
    _Float16* wfcT   = wprojT + (size_t)Dc * Dc;
    _Float16* wfc2T  = wfcT + (size_t)FFc * Dc;

    // attention branch
    layernorm_k<<<Mc, blk, 0, stream>>>(x, ln1g + (size_t)l * Dc,
                                        ln1b + (size_t)l * Dc, h16);
    gemm_wmma<ACT_NONE, false, true><<<dim3(3 * Dc / 128, MB), blk, 0, stream>>>(
        h16, wqkvT, bqkv + (size_t)l * 3 * Dc, nullptr, qkv16, Mc, 3 * Dc, Dc);
    attn_flash<<<Bc * Hc * (Tc / 128), blk, 0, stream>>>(qkv16, y16);
    gemm_wmma<ACT_NONE, false, false><<<dim3(Dc / 128, MB), blk, 0, stream>>>(
        y16, wprojT, bproj + (size_t)l * Dc, u, nullptr, Mc, Dc, Dc);

    // mlp branch
    layernorm_k<<<Mc, blk, 0, stream>>>(x, ln2g + (size_t)l * Dc,
                                        ln2b + (size_t)l * Dc, h16);
    gemm_wmma<ACT_GELU, false, true><<<dim3(FFc / 128, MB), blk, 0, stream>>>(
        h16, wfcT, bfc + (size_t)l * FFc, nullptr, hff16, Mc, FFc, Dc);
    gemm_wmma<ACT_NONE, true, false><<<dim3(Dc / 128, MB), blk, 0, stream>>>(
        hff16, wfc2T, bfc2 + (size_t)l * Dc, u, nullptr, Mc, Dc, FFc);

    // HORN momentum residual
    horn_k<<<(Mc * Dc) / 256, blk, 0, stream>>>(
        u, mlog + (size_t)l * Dc, etap + (size_t)l * Dc, v, x);
  }

  // final LN + tied LM head (Wtok16 already [V][D] == [N][K])
  layernorm_k<<<Mc, blk, 0, stream>>>(x, lnfg, lnfb, h16);
  gemm_wmma<ACT_NONE, false, false><<<dim3(Vc / 128, MB), blk, 0, stream>>>(
      h16, Wtok16, nullptr, out, nullptr, Mc, Vc, Dc);
}
